// CrossAttention_20160576487809
// MI455X (gfx1250) — matrix operand order repaired and checked by reading, not compiled, not run on hardware
//
#include <hip/hip_runtime.h>
#include <hip/hip_bf16.h>

// MI455X (gfx1250) cross-attention, bf16 WMMA pipeline with f32 accumulate.
// ~80 GFLOP, ~100MB HBM traffic at 23.3 TB/s -> bandwidth-dominated; bf16
// intermediates halve traffic, flash attention keeps scores on-chip, and the
// GEMM A-panel is fetched by the Tensor Data Mover (async, TENSORcnt-tracked).

#define SQ     2048
#define SKV    4096
#define DMODEL 768
#define NH     12
#define DHEAD  64
#define BATCH  2

typedef __bf16 bf16_t;
typedef __attribute__((ext_vector_type(16))) __bf16 v16bf;
typedef __attribute__((ext_vector_type(8)))  float  v8f;
typedef __attribute__((ext_vector_type(4)))  unsigned int u32x4;
typedef __attribute__((ext_vector_type(8)))  int i32x8;
typedef __attribute__((ext_vector_type(4)))  int i32x4;

#if defined(__has_builtin)
#if __has_builtin(__builtin_amdgcn_tensor_load_to_lds) && \
    __has_builtin(__builtin_amdgcn_s_wait_tensorcnt)
#define USE_TDM 1
#endif
#endif
#ifndef USE_TDM
#define USE_TDM 0
#endif

union Frag { v16bf v; unsigned int u[8]; };

__global__ void f32_to_bf16_kernel(const float* __restrict__ in,
                                   bf16_t* __restrict__ out, int n) {
    int i = blockIdx.x * blockDim.x + threadIdx.x;
    int stride = gridDim.x * blockDim.x;
    for (; i < n; i += stride) out[i] = (bf16_t)in[i];
}

__device__ inline float half_reduce_max(float x) {
    x = fmaxf(x, __shfl_xor(x, 1, 32));
    x = fmaxf(x, __shfl_xor(x, 2, 32));
    x = fmaxf(x, __shfl_xor(x, 4, 32));
    x = fmaxf(x, __shfl_xor(x, 8, 32));
    return x;
}
__device__ inline float half_reduce_sum(float x) {
    x += __shfl_xor(x, 1, 32);
    x += __shfl_xor(x, 2, 32);
    x += __shfl_xor(x, 4, 32);
    x += __shfl_xor(x, 8, 32);
    return x;
}

// C = A[M,K] * B[K,N] (+bias), bf16 in, f32 accumulate via v_wmma_f32_16x16x32_bf16.
// Block = 256 threads = 8 waves. Tile: 128(M) x 64(N), k-step 32.
// A-panel (128x32 bf16, dense rows of 64B) loaded by TDM when available.
// MODE 0: bf16 out, (acc+bias)*scale         (Q/K projections; Q folds 1/sqrt(64))
// MODE 1: bf16 out transposed V^T [B,H,DH,S] (V projection)
// MODE 2: f32 out, acc+bias                  (output projection)
template<int MODE>
__global__ __launch_bounds__(256) void gemm_bf16_wmma(
    const bf16_t* __restrict__ A, const bf16_t* __restrict__ Bm,
    const float* __restrict__ bias, void* __restrict__ outp,
    int M, int N, int K, float scale, int Srows)
{
    __shared__ __align__(16) bf16_t lA[128 * 32];   // [m][k] row-major
    __shared__ __align__(16) bf16_t lB[64 * 32];    // [n][k] (transposed on fill)
    int tid = threadIdx.x, lane = tid & 31, wv = tid >> 5;
    int half = lane >> 4, nl = lane & 15;
    int m0 = blockIdx.y * 128, n0 = blockIdx.x * 64;

    v8f acc[4] = {};

    for (int k0 = 0; k0 < K; k0 += 32) {
        __syncthreads();
#if USE_TDM
        if (wv == 0) {
            // 2D TDM descriptor: tile 32(K) x 128(M) bf16 from row-major [M,K],
            // landing dense at lA. One wave issues; TENSORcnt gates completion.
            unsigned lds_byte = (unsigned)(size_t)(void*)lA;  // low 32b = LDS offset
            unsigned long long ga =
                (unsigned long long)(size_t)(A + (size_t)m0 * K + k0);
            u32x4 g0;
            g0[0] = 1u;                                   // count=1, user mode
            g0[1] = lds_byte;                             // lds_addr (bytes)
            g0[2] = (unsigned)(ga & 0xFFFFFFFFu);         // global_addr[31:0]
            g0[3] = (unsigned)((ga >> 32) & 0x01FFFFFFu)  // global_addr[56:32]
                  | (2u << 30);                           // type=2 (image)
            i32x8 g1;
            g1[0] = (int)(1u << 16);                      // data_size=1 (2 bytes)
            g1[1] = (int)(((unsigned)K & 0xFFFFu) << 16); // tensor_dim0[15:0]
            g1[2] = (int)(((unsigned)K >> 16)             // tensor_dim0[31:16]
                  | (((unsigned)M & 0xFFFFu) << 16));     // tensor_dim1[15:0]
            g1[3] = (int)(((unsigned)M >> 16)             // tensor_dim1[31:16]
                  | (32u << 16));                         // tile_dim0 = 32
            g1[4] = (int)128u;                            // tile_dim1 = 128
            g1[5] = (int)(unsigned)K;                     // tensor_dim0_stride
            g1[6] = 0;
            g1[7] = 0;
            i32x4 z4 = {0, 0, 0, 0};                      // <=2D: groups 2/3 unused
            i32x8 z8 = {0, 0, 0, 0, 0, 0, 0, 0};
            __builtin_amdgcn_tensor_load_to_lds(g0, g1, z4, z4, z8, 0);
            __builtin_amdgcn_s_wait_tensorcnt(0);
        }
#else
        {   // A tile: 128x32 bf16 = 8KB; each thread copies 32B
            int r = tid >> 1, p = tid & 1;
            const uint4* src = (const uint4*)(A + (size_t)(m0 + r) * K + k0 + p * 16);
            uint4* dst = (uint4*)(lA + r * 32 + p * 16);
            dst[0] = src[0];
            dst[1] = src[1];
        }
#endif
        {   // B tile: 32x64 -> store transposed [n][k]
            int k = tid & 31, ng = tid >> 5;
            uint4 d4 = *(const uint4*)(Bm + (size_t)(k0 + k) * N + n0 + ng * 8);
            const bf16_t* dv = (const bf16_t*)&d4;
            #pragma unroll
            for (int i = 0; i < 8; ++i) lB[(ng * 8 + i) * 32 + k] = dv[i];
        }
        __syncthreads();
        if (k0 + 32 < K)
            __builtin_prefetch(Bm + (size_t)(k0 + 32 + (tid & 31)) * N + n0, 0, 0);

        const unsigned int* lAu = (const unsigned int*)lA;
        const unsigned int* lBu = (const unsigned int*)lB;
        Frag a;
        int mrow = wv * 16 + nl;
        #pragma unroll
        for (int jj = 0; jj < 8; ++jj) {   // A frag: K pairs {2j+8h} / {2(j-4)+16+8h}
            int idx = (jj < 4) ? (jj + 4 * half) : (jj + 4 + 4 * half);
            a.u[jj] = lAu[mrow * 16 + idx];
        }
        #pragma unroll
        for (int t = 0; t < 4; ++t) {
            Frag b;
            #pragma unroll
            for (int jj = 0; jj < 8; ++jj)  // B frag: K pairs {2j+8h} / {2(j-4)+16+8h} at column n
                b.u[jj] = lBu[(t * 16 + nl) * 16 + ((jj < 4) ? (jj + 4 * half) : (jj + 4 + 4 * half))];
            acc[t] = __builtin_amdgcn_wmma_f32_16x16x32_bf16(
                false, a.v, false, b.v, (short)0, acc[t], false, false);
        }
    }

    #pragma unroll
    for (int t = 0; t < 4; ++t) {
        #pragma unroll
        for (int v = 0; v < 8; ++v) {       // C layout: row = v + 8*half, col = nl
            int gm = m0 + wv * 16 + v + 8 * half;
            int gn = n0 + t * 16 + nl;
            float val = acc[t][v] + bias[gn];
            if (MODE == 0) {
                ((bf16_t*)outp)[(size_t)gm * N + gn] = (bf16_t)(val * scale);
            } else if (MODE == 1) {
                int b = gm / Srows, s = gm % Srows;
                int h = gn / DHEAD, dh = gn % DHEAD;
                ((bf16_t*)outp)[(((size_t)(b * NH + h)) * DHEAD + dh) * Srows + s] = (bf16_t)val;
            } else {
                ((float*)outp)[(size_t)gm * N + gn] = val;
            }
        }
    }
}

// Flash attention. Q pre-scaled by 1/8. 256 threads = 8 independent waves;
// each wave owns 16 q rows, streams kv in 64-column tiles (16 WMMAs per
// softmax-reduction pass to amortize the shuffle/exp overhead).
__global__ __launch_bounds__(256) void attn_wmma(
    const bf16_t* __restrict__ Q, const bf16_t* __restrict__ Kc,
    const bf16_t* __restrict__ Vt, bf16_t* __restrict__ ctx)
{
    __shared__ __align__(16) bf16_t lP[8][16 * 64];  // per-wave P tile (C->A relayout)
    int tid = threadIdx.x, lane = tid & 31, wv = tid >> 5;
    int half = lane >> 4, nl = lane & 15;
    int qt = blockIdx.x & 15;           // SQ/128 = 16 q-tiles
    int bh = blockIdx.x >> 4;
    int h = bh % NH, b = bh / NH;
    int q0 = qt * 128 + wv * 16;

    Frag qa[2];                          // Q rows, A-fragments for DH blocks 0,32
    #pragma unroll
    for (int d = 0; d < 2; ++d) {
        #pragma unroll
        for (int jj = 0; jj < 8; ++jj) {
            int koff = (jj < 4) ? (2 * jj + 8 * half) : (2 * (jj - 4) + 16 + 8 * half);
            qa[d].u[jj] = *(const unsigned int*)(
                Q + (size_t)(b * SQ + q0 + nl) * DMODEL + h * DHEAD + d * 32 + koff);
        }
    }

    float mrow[8], srow[8];
    v8f o[4] = {};
    #pragma unroll
    for (int v = 0; v < 8; ++v) { mrow[v] = -3.0e38f; srow[v] = 0.f; }

    bf16_t* lp = lP[wv];
    const unsigned int* lpu = (const unsigned int*)lp;

    for (int kv = 0; kv < SKV; kv += 64) {
        v8f s[4] = {};
        #pragma unroll
        for (int d = 0; d < 2; ++d) {    // scores: S[16q x 64kv] = Q * K^T
            #pragma unroll
            for (int t = 0; t < 4; ++t) {
                Frag kb;                 // B frag: pairs along DH contiguous in K
                #pragma unroll
                for (int jj = 0; jj < 8; ++jj) {
                    int koff = h * DHEAD + d * 32 + ((jj < 4) ? (2 * jj + 8 * half) : (2 * (jj - 4) + 16 + 8 * half));
                    kb.u[jj] = *(const unsigned int*)(
                        Kc + (size_t)(b * SKV + kv + t * 16 + nl) * DMODEL + koff);
                }
                s[t] = __builtin_amdgcn_wmma_f32_16x16x32_bf16(
                    false, qa[d].v, false, kb.v, (short)0, s[t], false, false);
            }
        }
        // online softmax; each 16-lane half shares the same 8 rows
        float p[4][8], alpha[8];
        #pragma unroll
        for (int v = 0; v < 8; ++v) {
            float r = fmaxf(fmaxf(s[0][v], s[1][v]), fmaxf(s[2][v], s[3][v]));
            r = half_reduce_max(r);
            float mn = fmaxf(mrow[v], r);
            alpha[v] = __expf(mrow[v] - mn);
            float rs = 0.f;
            #pragma unroll
            for (int t = 0; t < 4; ++t) { p[t][v] = __expf(s[t][v] - mn); rs += p[t][v]; }
            rs = half_reduce_sum(rs);
            srow[v] = srow[v] * alpha[v] + rs;
            mrow[v] = mn;
        }
        #pragma unroll
        for (int t = 0; t < 4; ++t)
            #pragma unroll
            for (int v = 0; v < 8; ++v) o[t][v] = o[t][v] * alpha[v];

        // P: C-layout -> A-fragment via per-wave LDS round-trip
        #pragma unroll
        for (int t = 0; t < 4; ++t)
            #pragma unroll
            for (int v = 0; v < 8; ++v)
                lp[(v + 8 * half) * 64 + t * 16 + nl] = (bf16_t)p[t][v];
        asm volatile("s_wait_dscnt 0" ::: "memory");   // CDNA5 split DS counter wait
        Frag pa[2];
        #pragma unroll
        for (int c = 0; c < 2; ++c)
            #pragma unroll
            for (int jj = 0; jj < 8; ++jj) {
                int idx = (jj < 4) ? (jj + 4 * half) : (jj + 4 + 4 * half);
                pa[c].u[jj] = lpu[nl * 32 + c * 16 + idx];
            }
        #pragma unroll
        for (int t = 0; t < 4; ++t) {    // O += P * V ; V^T makes kv pairs contiguous
            #pragma unroll
            for (int c = 0; c < 2; ++c) {
                Frag vb;
                #pragma unroll
                for (int jj = 0; jj < 8; ++jj)
                    vb.u[jj] = *(const unsigned int*)(
                        Vt + ((size_t)((b * NH + h) * DHEAD + t * 16 + nl)) * SKV
                           + kv + c * 32 + ((jj < 4) ? (2 * jj + 8 * half) : (2 * (jj - 4) + 16 + 8 * half)));
                o[t] = __builtin_amdgcn_wmma_f32_16x16x32_bf16(
                    false, pa[c].v, false, vb.v, (short)0, o[t], false, false);
            }
        }
    }

    float inv[8];
    #pragma unroll
    for (int v = 0; v < 8; ++v) inv[v] = 1.0f / srow[v];
    #pragma unroll
    for (int t = 0; t < 4; ++t)
        #pragma unroll
        for (int v = 0; v < 8; ++v)
            ctx[(size_t)(b * SQ + q0 + v + 8 * half) * DMODEL + h * DHEAD + t * 16 + nl]
                = (bf16_t)(o[t][v] * inv[v]);
}

extern "C" void kernel_launch(void* const* d_in, const int* in_sizes, int n_in,
                              void* d_out, int out_size, void* d_ws, size_t ws_size,
                              hipStream_t stream) {
    const float* hs  = (const float*)d_in[0];
    const float* enc = (const float*)d_in[1];
    const float* Wq  = (const float*)d_in[2];
    const float* bq  = (const float*)d_in[3];
    const float* Wk  = (const float*)d_in[4];
    const float* bk  = (const float*)d_in[5];
    const float* Wv  = (const float*)d_in[6];
    const float* bv  = (const float*)d_in[7];
    const float* Wo  = (const float*)d_in[8];
    const float* bo  = (const float*)d_in[9];
    float* out = (float*)d_out;

    char* ws = (char*)d_ws;
    size_t off = 0;
    auto alloc = [&](size_t bytes) -> void* {
        void* p = ws + off;
        off += bytes;
        off = (off + 255) & ~(size_t)255;
        return p;
    };
    const size_t nHS  = (size_t)BATCH * SQ * DMODEL;     // 3,145,728
    const size_t nENC = (size_t)BATCH * SKV * DMODEL;    // 6,291,456
    const size_t nW   = (size_t)DMODEL * DMODEL;

    bf16_t* hs_bf  = (bf16_t*)alloc(nHS * 2);
    bf16_t* enc_bf = (bf16_t*)alloc(nENC * 2);
    bf16_t* wq_bf  = (bf16_t*)alloc(nW * 2);
    bf16_t* wk_bf  = (bf16_t*)alloc(nW * 2);
    bf16_t* wv_bf  = (bf16_t*)alloc(nW * 2);
    bf16_t* wo_bf  = (bf16_t*)alloc(nW * 2);
    bf16_t* q_bf   = (bf16_t*)alloc(nHS * 2);            // [B*Sq, 768], pre-scaled 1/8
    bf16_t* k_bf   = (bf16_t*)alloc(nENC * 2);           // [B*Skv, 768]
    bf16_t* vt_bf  = (bf16_t*)alloc(nENC * 2);           // [B,H,DH,Skv]
    bf16_t* ctx_bf = (bf16_t*)alloc(nHS * 2);            // [B*Sq, 768]

    f32_to_bf16_kernel<<<1024, 256, 0, stream>>>(hs,  hs_bf,  (int)nHS);
    f32_to_bf16_kernel<<<2048, 256, 0, stream>>>(enc, enc_bf, (int)nENC);
    f32_to_bf16_kernel<<<512,  256, 0, stream>>>(Wq,  wq_bf,  (int)nW);
    f32_to_bf16_kernel<<<512,  256, 0, stream>>>(Wk,  wk_bf,  (int)nW);
    f32_to_bf16_kernel<<<512,  256, 0, stream>>>(Wv,  wv_bf,  (int)nW);
    f32_to_bf16_kernel<<<512,  256, 0, stream>>>(Wo,  wo_bf,  (int)nW);

    dim3 blk(256);
    // Q = (hs @ Wq + bq) * 0.125
    gemm_bf16_wmma<0><<<dim3(DMODEL / 64, (BATCH * SQ) / 128), blk, 0, stream>>>(
        hs_bf, wq_bf, bq, q_bf, BATCH * SQ, DMODEL, DMODEL, 0.125f, SQ);
    // K = enc @ Wk + bk
    gemm_bf16_wmma<0><<<dim3(DMODEL / 64, (BATCH * SKV) / 128), blk, 0, stream>>>(
        enc_bf, wk_bf, bk, k_bf, BATCH * SKV, DMODEL, DMODEL, 1.0f, SKV);
    // V^T = transpose_heads(enc @ Wv + bv)
    gemm_bf16_wmma<1><<<dim3(DMODEL / 64, (BATCH * SKV) / 128), blk, 0, stream>>>(
        enc_bf, wv_bf, bv, vt_bf, BATCH * SKV, DMODEL, DMODEL, 1.0f, SKV);
    // ctx = softmax(Q K^T) V
    attn_wmma<<<BATCH * NH * (SQ / 128), blk, 0, stream>>>(q_bf, k_bf, vt_bf, ctx_bf);
    // out = ctx @ Wo + bo  (f32)
    gemm_bf16_wmma<2><<<dim3(DMODEL / 64, (BATCH * SQ) / 128), blk, 0, stream>>>(
        ctx_bf, wo_bf, bo, out, BATCH * SQ, DMODEL, DMODEL, 1.0f, SQ);
}
